// AttentionZP_26766236189060
// MI455X (gfx1250) — matrix operand order repaired and checked by reading, not compiled, not run on hardware
//
#include <hip/hip_runtime.h>
#include <hip/hip_bf16.h>

typedef __attribute__((ext_vector_type(16))) _Float16 v16h;
typedef __attribute__((ext_vector_type(8)))  _Float16 v8h;
typedef __attribute__((ext_vector_type(4)))  _Float16 v4h;
typedef __attribute__((ext_vector_type(8)))  float    v8f;
typedef __attribute__((ext_vector_type(4)))  unsigned int u32x4;
typedef __attribute__((ext_vector_type(8)))  int      i32x8;
typedef __attribute__((ext_vector_type(4)))  int      i32x4;

// problem dims
static constexpr int Hh = 8, Tt = 1024, Aa = 64, Cc = 256;
static constexpr int Bb = 2, Qq = 2048, Kk = 2048;
static constexpr int HC = Hh * Cc; // 2048

#define DEVFN __device__ __forceinline__

// ---------------------------------------------------------------------------
// WMMA helpers (layouts per CDNA5 ISA, wave32)
// ---------------------------------------------------------------------------

// A-tile 16x32 f16: lane l holds M=l&15, elems 0..7 = K akb..akb+7,
// elems 8..15 = K akb+16..akb+23 (akb pre-folded into p by caller)
DEVFN v16h load_a16(const _Float16* __restrict__ p) {
    v8h lo = *(const v8h*)(p);
    v8h hi = *(const v8h*)(p + 16);
    return __builtin_shufflevector(lo, hi, 0,1,2,3,4,5,6,7,8,9,10,11,12,13,14,15);
}

DEVFN v8f wmma_f16(v16h a, v16h b, v8f c) {
    return __builtin_amdgcn_wmma_f32_16x16x32_f16(false, a, false, b, (short)0, c,
                                                  false, false);
}

// One wave computes a 32 x (16*NT) f32 block of D = A(MxK) * Bt(NxK)^T.
// Both operands K-contiguous -> global_load_b128 pairs per lane.
template <int NT>
DEVFN void gemm_nt_wave(const _Float16* __restrict__ A, int lda,
                        const _Float16* __restrict__ Bt, int ldb,
                        int Kd, int m0, int n0, v8f acc[2][NT]) {
    const int lane = threadIdx.x & 31;
    const int r    = lane & 15;
    const int hi   = lane >> 4;
    const int akb  = hi ? 8 : 0;    // A lane-half K offset
    const int bkb  = hi ? 8 : 0;    // B lane-half K offset
    const _Float16* a0p = A + (size_t)(m0 + r)      * lda + akb;
    const _Float16* a1p = A + (size_t)(m0 + 16 + r) * lda + akb;
    const _Float16* bp[NT];
#pragma unroll
    for (int j = 0; j < NT; ++j)
        bp[j] = Bt + (size_t)(n0 + 16 * j + r) * ldb + bkb;
    for (int k0 = 0; k0 < Kd; k0 += 32) {
        v16h a0 = load_a16(a0p + k0);
        v16h a1 = load_a16(a1p + k0);
        v16h b[NT];
#pragma unroll
        for (int j = 0; j < NT; ++j) b[j] = load_a16(bp[j] + k0);
#pragma unroll
        for (int j = 0; j < NT; ++j) {
            acc[0][j] = wmma_f16(a0, b[j], acc[0][j]);
            acc[1][j] = wmma_f16(a1, b[j], acc[1][j]);
        }
    }
}

// D layout: lane l: N = l&15; VGPR v holds M = v + 8*(l>=16)
template <typename T, int NT>
DEVFN void store_acc(T* __restrict__ D, int ldd, int m0, int n0,
                     const v8f acc[2][NT], const float* __restrict__ bias) {
    const int lane   = threadIdx.x & 31;
    const int col    = lane & 15;
    const int rowoff = (lane >> 4) * 8;
#pragma unroll
    for (int i = 0; i < 2; ++i)
#pragma unroll
        for (int j = 0; j < NT; ++j) {
            const int n  = n0 + j * 16 + col;
            const float bv = bias ? bias[n] : 0.0f;
#pragma unroll
            for (int v = 0; v < 8; ++v) {
                const int m = m0 + i * 16 + rowoff + v;
                D[(size_t)m * ldd + n] = (T)(acc[i][j][v] + bv);
            }
        }
}

// 8 waves: 4 (M) x 2 (N); WG tile = 128 x (32*NT)
template <int NT>
DEVFN void wave_tile_origin(int& m0, int& n0) {
    const int wave = threadIdx.x >> 5;
    m0 = blockIdx.y * 128 + (wave >> 1) * 32;
    n0 = blockIdx.x * (32 * NT) + (wave & 1) * (16 * NT);
}

// ---------------------------------------------------------------------------
// GEMM wrapper kernels (all NT-form, f16 in, f32 accumulate)
// ---------------------------------------------------------------------------

// proj: [2048 x T] x [T x A] per (b,h); N=64 -> 32x32 wave tile
__global__ __launch_bounds__(256) void k_gemm_proj(
    const _Float16* __restrict__ tok16,   // [B, 2048, T]
    const _Float16* __restrict__ wT,      // [H, A, T]
    const float* __restrict__ bias,       // [H, A] or null
    _Float16* __restrict__ out) {         // [B*H, 2048, A]
    const int z = blockIdx.z, b = z / Hh, h = z % Hh;
    int m0, n0; wave_tile_origin<2>(m0, n0);
    v8f acc[2][2] = {};
    gemm_nt_wave<2>(tok16 + (size_t)b * Kk * Tt, Tt,
                    wT    + (size_t)h * Aa * Tt, Tt, Tt, m0, n0, acc);
    store_acc<_Float16, 2>(out + (size_t)z * Kk * Aa, Aa, m0, n0, acc,
                           bias ? bias + h * Aa : nullptr);
}

// logits[k,q] = kproj[k,:] . qproj[q,:]
__global__ __launch_bounds__(256) void k_gemm_logits(
    const _Float16* __restrict__ kproj, const _Float16* __restrict__ qproj,
    float* __restrict__ logits) {
    const int z = blockIdx.z;
    int m0, n0; wave_tile_origin<4>(m0, n0);
    v8f acc[2][4] = {};
    gemm_nt_wave<4>(kproj + (size_t)z * Kk * Aa, Aa,
                    qproj + (size_t)z * Qq * Aa, Aa, Aa, m0, n0, acc);
    store_acc<float, 4>(logits + (size_t)z * Kk * Qq, Qq, m0, n0, acc, nullptr);
}

// vs[q,t] = sum_k attnT[q,k] * keyT[t,k]   (dominant GEMM, 137 GFLOP)
__global__ __launch_bounds__(256) void k_gemm_vs(
    const _Float16* __restrict__ attnT,   // [B*H, Q, K]
    const _Float16* __restrict__ keyT,    // [B, T, K]
    _Float16* __restrict__ vs16) {        // [B*H, Q, T]
    const int z = blockIdx.z, b = z / Hh;
    int m0, n0; wave_tile_origin<4>(m0, n0);
    v8f acc[2][4] = {};
    gemm_nt_wave<4>(attnT + (size_t)z * Qq * Kk, Kk,
                    keyT  + (size_t)b * Tt * Kk, Kk, Kk, m0, n0, acc);
    store_acc<_Float16, 4>(vs16 + (size_t)z * Qq * Tt, Tt, m0, n0, acc, nullptr);
}

// vsc[b,q,h*C+c] = sum_t vs[q,t] * vdT[c,t]   (head interleave -> free head-sum)
__global__ __launch_bounds__(256) void k_gemm_vsc(
    const _Float16* __restrict__ vs16,    // [B*H, Q, T]
    const _Float16* __restrict__ vdT,     // [H, C, T]
    _Float16* __restrict__ vsc) {         // [B, Q, H*C]
    const int z = blockIdx.z, b = z / Hh, h = z % Hh;
    int m0, n0; wave_tile_origin<4>(m0, n0);
    v8f acc[2][4] = {};
    gemm_nt_wave<4>(vs16 + (size_t)z * Qq * Tt, Tt,
                    vdT  + (size_t)h * Cc * Tt, Tt, Tt, m0, n0, acc);
    store_acc<_Float16, 4>(vsc + (size_t)b * Qq * HC + h * Cc, HC, m0, n0, acc, nullptr);
}

// out[b,q,t] = sum_{h,c} vsc[b,q,hc] * vupT[t,hc]  (K-dim = H*C sums heads)
__global__ __launch_bounds__(256) void k_gemm_out(
    const _Float16* __restrict__ vsc, const _Float16* __restrict__ vupT,
    float* __restrict__ out) {
    const int b = blockIdx.z;
    int m0, n0; wave_tile_origin<4>(m0, n0);
    v8f acc[2][4] = {};
    gemm_nt_wave<4>(vsc + (size_t)b * Qq * HC, HC, vupT, HC, HC, m0, n0, acc);
    store_acc<float, 4>(out + (size_t)b * Qq * Tt, Tt, m0, n0, acc, nullptr);
}

// ---------------------------------------------------------------------------
// swishmax pieces
// ---------------------------------------------------------------------------

// Online max / abs-sum over the K axis, 4 interleaved accumulators for ILP.
__global__ __launch_bounds__(256) void k_swish_reduce(
    const float* __restrict__ logits, float* __restrict__ mbuf,
    float* __restrict__ sbuf) {
    const int z = blockIdx.z;
    const int q = blockIdx.x * 256 + threadIdx.x;
    const float* p = logits + (size_t)z * Kk * Qq + q;
    float m[4] = {-3.0e38f, -3.0e38f, -3.0e38f, -3.0e38f};
    float d[4] = {0.0f, 0.0f, 0.0f, 0.0f};
    for (int k = 0; k < Kk; k += 4) {
#pragma unroll
        for (int j = 0; j < 4; ++j) {
            const float x  = p[(size_t)(k + j) * Qq];
            const float mn = fmaxf(m[j], x);
            d[j] = d[j] * __expf(m[j] - mn) + fabsf(x) * __expf(x - mn);
            m[j] = mn;
        }
    }
    const float M = fmaxf(fmaxf(m[0], m[1]), fmaxf(m[2], m[3]));
    const float D = d[0] * __expf(m[0] - M) + d[1] * __expf(m[1] - M) +
                    d[2] * __expf(m[2] - M) + d[3] * __expf(m[3] - M);
    mbuf[(size_t)z * Qq + q] = M;
    sbuf[(size_t)z * Qq + q] = 1.0f / (D + 1.0f);  // NOT_EPSILON = 1.0
}

// attnT[q,k] = f16( x * exp(x - m_q) * s_q ), LDS-tiled transpose.
// Tile fetch uses the Tensor Data Mover when available: one wave issues a 2D
// tensor_load_to_lds whose pad fields (32 DWORDs interval, 1 DWORD pad) realize
// the bank-conflict-free tile[32][33] layout in hardware.
__global__ void k_attn_tr(const float* __restrict__ logits,
                          const float* __restrict__ mbuf,
                          const float* __restrict__ sbuf,
                          _Float16* __restrict__ attnT) {
    __shared__ float tile[32][33];
    const int z  = blockIdx.z;
    const int k0 = blockIdx.y * 32, q0 = blockIdx.x * 32;
    const int tx = threadIdx.x, ty = threadIdx.y;
    const float* src = logits + (size_t)z * Kk * Qq;

#if __has_builtin(__builtin_amdgcn_tensor_load_to_lds) && \
    __has_builtin(__builtin_amdgcn_s_wait_tensorcnt)
    if (ty == 0) {  // first wave of the block issues the DMA (EXEC ignored)
        const unsigned int ldsa = (unsigned int)(uintptr_t)&tile[0][0];
        const unsigned long long ga =
            (unsigned long long)(uintptr_t)(src + (size_t)k0 * Qq + q0);
        u32x4 g0;
        g0[0] = 1u;                         // count=1, user mode, no gather
        g0[1] = ldsa;                       // lds_addr
        g0[2] = (unsigned int)ga;           // global_addr[31:0]
        g0[3] = (unsigned int)((ga >> 32) & 0x01FFFFFFu) | (2u << 30); // type=2
        i32x8 g1;
        g1[0] = (int)((2u << 16) |          // data_size = 4B
                      (1u << 20) |          // pad_enable
                      (4u << 22));          // pad_interval: 32 DWORDs (+1 DWORD)
        g1[1] = (int)(((unsigned)Qq & 0xFFFFu) << 16);  // tensor_dim0 lo16 in [63:48]
        g1[2] = (int)(((unsigned)Qq & 0xFFFFu) << 16);  // dim0 hi=0 | tensor_dim1 lo16
        g1[3] = (int)(32u << 16);           // tensor_dim1 hi=0 | tile_dim0 = 32
        g1[4] = (int)32u;                   // tile_dim1 = 32, tile_dim2 = 0
        g1[5] = (int)(unsigned)Qq;          // tensor_dim0_stride[31:0]
        g1[6] = 0;                          // stride hi / tensor_dim1_stride lo
        g1[7] = 0;
        i32x4 gz = {};
        i32x8 gz8 = {};
        // 6-arg toolchain form: (g0, g1, g2, g3, g4, cpol)
        __builtin_amdgcn_tensor_load_to_lds(g0, g1, gz, gz, gz8, 0);
        __builtin_amdgcn_s_wait_tensorcnt(0);
    }
    __syncthreads();
#else
#pragma unroll
    for (int i = 0; i < 4; ++i)
        tile[ty + i * 8][tx] = src[(size_t)(k0 + ty + i * 8) * Qq + q0 + tx];
    __syncthreads();
#endif

    _Float16* dst = attnT + (size_t)z * Qq * Kk;
#pragma unroll
    for (int i = 0; i < 4; ++i) {
        const int q    = q0 + ty + i * 8;
        const float mq = mbuf[(size_t)z * Qq + q];
        const float sq = sbuf[(size_t)z * Qq + q];
        const float x  = tile[tx][ty + i * 8];
        dst[(size_t)q * Kk + k0 + tx] = (_Float16)(x * __expf(x - mq) * sq);
    }
}

// ---------------------------------------------------------------------------
// layout prep: f32 -> f16 converts and transposes
// ---------------------------------------------------------------------------

__global__ void k_cvt_f16(const float* __restrict__ src,
                          _Float16* __restrict__ dst, long n4) {
    const long i = (long)blockIdx.x * blockDim.x + threadIdx.x;
    if (i < n4) {
        const float4 f = ((const float4*)src)[i];
        v4h h;
        h[0] = (_Float16)f.x; h[1] = (_Float16)f.y;
        h[2] = (_Float16)f.z; h[3] = (_Float16)f.w;
        ((v4h*)dst)[i] = h;
    }
}

// src: batch z of [R, Cd] f32 row-major -> dst: batch z of [Cd, R] f16 row-major
__global__ void k_transpose_cvt(const float* __restrict__ src,
                                _Float16* __restrict__ dst, int R, int Cd) {
    __shared__ float tile[32][33];
    const size_t base = (size_t)blockIdx.z * R * Cd;
    const int r0 = blockIdx.y * 32, c0 = blockIdx.x * 32;
    const int tx = threadIdx.x, ty = threadIdx.y;
#pragma unroll
    for (int i = 0; i < 4; ++i)
        tile[ty + i * 8][tx] = src[base + (size_t)(r0 + ty + i * 8) * Cd + c0 + tx];
    __syncthreads();
#pragma unroll
    for (int i = 0; i < 4; ++i)
        dst[base + (size_t)(c0 + ty + i * 8) * R + r0 + tx] =
            (_Float16)tile[tx][ty + i * 8];
}

// ---------------------------------------------------------------------------
// launch
// ---------------------------------------------------------------------------

extern "C" void kernel_launch(void* const* d_in, const int* in_sizes, int n_in,
                              void* d_out, int out_size, void* d_ws, size_t ws_size,
                              hipStream_t stream) {
    (void)in_sizes; (void)n_in; (void)out_size; (void)ws_size;
    const float* query_tokens = (const float*)d_in[0];  // [B,Q,T]
    const float* key_tokens   = (const float*)d_in[1];  // [B,K,T]
    const float* key_down     = (const float*)d_in[2];  // [H,T,A]
    const float* query_down   = (const float*)d_in[3];  // [H,T,A]
    const float* qdb          = (const float*)d_in[4];  // [H,1,A]
    const float* value_down   = (const float*)d_in[5];  // [H,T,C]
    const float* value_up     = (const float*)d_in[6];  // [H,C,T]

    // bump allocator in workspace (256B aligned so v16h/v8h loads stay aligned)
    size_t off = 0;
    auto alloc = [&](size_t nbytes) -> void* {
        off = (off + 255) & ~(size_t)255;
        void* p = (char*)d_ws + off;
        off += nbytes;
        return p;
    };
    _Float16* key16 = (_Float16*)alloc((size_t)Bb * Kk * Tt * 2);
    _Float16* q16   = (_Float16*)alloc((size_t)Bb * Qq * Tt * 2);
    _Float16* keyT  = (_Float16*)alloc((size_t)Bb * Tt * Kk * 2);
    _Float16* kdT   = (_Float16*)alloc((size_t)Hh * Aa * Tt * 2);
    _Float16* qdT   = (_Float16*)alloc((size_t)Hh * Aa * Tt * 2);
    _Float16* vdT   = (_Float16*)alloc((size_t)Hh * Cc * Tt * 2);
    _Float16* vupT  = (_Float16*)alloc((size_t)Tt * HC * 2);
    _Float16* kproj = (_Float16*)alloc((size_t)Bb * Hh * Kk * Aa * 2);
    _Float16* qproj = (_Float16*)alloc((size_t)Bb * Hh * Qq * Aa * 2);
    float*    mbuf  = (float*)alloc((size_t)Bb * Hh * Qq * 4);
    float*    sbuf  = (float*)alloc((size_t)Bb * Hh * Qq * 4);
    float*    logits= (float*)alloc((size_t)Bb * Hh * Kk * Qq * 4);    // 268 MB
    _Float16* attnT = (_Float16*)alloc((size_t)Bb * Hh * Qq * Kk * 2); // 134 MB
    // vs/vsc alias the logits region (logits dead once attnT is written)
    _Float16* vs16  = (_Float16*)logits;                        // 67 MB
    _Float16* vsc   = vs16 + (size_t)Bb * Hh * Qq * Tt;         // 16.8 MB

    const dim3 blk(256);
    const dim3 tb(32, 8);

    // layout prep
    const long ntok4 = (long)Bb * Kk * Tt / 4;
    k_cvt_f16<<<dim3((unsigned)((ntok4 + 255) / 256)), blk, 0, stream>>>(key_tokens, key16, ntok4);
    k_cvt_f16<<<dim3((unsigned)((ntok4 + 255) / 256)), blk, 0, stream>>>(query_tokens, q16, ntok4);
    k_transpose_cvt<<<dim3(Aa / 32, Tt / 32, Hh), tb, 0, stream>>>(key_down,   kdT, Tt, Aa);
    k_transpose_cvt<<<dim3(Aa / 32, Tt / 32, Hh), tb, 0, stream>>>(query_down, qdT, Tt, Aa);
    k_transpose_cvt<<<dim3(Cc / 32, Tt / 32, Hh), tb, 0, stream>>>(value_down, vdT, Tt, Cc);
    k_transpose_cvt<<<dim3(Tt / 32, HC / 32, 1),  tb, 0, stream>>>(value_up,   vupT, HC, Tt);
    k_transpose_cvt<<<dim3(Tt / 32, Kk / 32, Bb), tb, 0, stream>>>(key_tokens, keyT, Kk, Tt);

    // projections (WMMA)
    k_gemm_proj<<<dim3(Aa / 64, Kk / 128, Bb * Hh), blk, 0, stream>>>(key16, kdT, nullptr, kproj);
    k_gemm_proj<<<dim3(Aa / 64, Qq / 128, Bb * Hh), blk, 0, stream>>>(q16, qdT, qdb, qproj);

    // logits (WMMA) + swishmax (online reduce, then fused transform+transpose)
    k_gemm_logits<<<dim3(Qq / 128, Kk / 128, Bb * Hh), blk, 0, stream>>>(kproj, qproj, logits);
    k_swish_reduce<<<dim3(Qq / 256, 1, Bb * Hh), blk, 0, stream>>>(logits, mbuf, sbuf);
    k_attn_tr<<<dim3(Qq / 32, Kk / 32, Bb * Hh), tb, 0, stream>>>(logits, mbuf, sbuf, attnT);

    // attn^T x key (dominant GEMM), compress down, compress up + head-sum
    k_gemm_vs<<<dim3(Tt / 128, Qq / 128, Bb * Hh), blk, 0, stream>>>(attnT, keyT, vs16);
    k_gemm_vsc<<<dim3(Cc / 128, Qq / 128, Bb * Hh), blk, 0, stream>>>(vs16, vdT, vsc);
    k_gemm_out<<<dim3(Tt / 128, Qq / 128, Bb), blk, 0, stream>>>(vsc, vupT, (float*)d_out);
}
